// QuantizedLinear_55009941127907
// MI455X (gfx1250) — matrix operand order repaired and checked by reading, not compiled, not run on hardware
//
#include <hip/hip_runtime.h>

// ---------------------------------------------------------------------------
// Quantized (int4, group-32) linear:  y[M,OUT] = x[M,IN] @ W[OUT,IN]^T + bias
// CDNA5 / gfx1250: f16 WMMA (V_WMMA_F32_16X16X32_F16), wave32, LDS tiling.
//   - workgroup: 256 threads (8 waves), output tile 128(M) x 128(N)
//   - K chunk:   64 (two K=32 WMMA steps), double-buffered LDS
//   - dequant:   packed-f16 magic (0x6400|q = 1024+q), exact pk_add, pk_mul
//   - per-wave:  64x32 sub-tile -> 4x2 accumulators of 16x16 f32
// ---------------------------------------------------------------------------

#define TM 128
#define TN 128
#define TK 64
#define XP 72   // padded LDS row (elements): 144B = 9*16B, keeps b128 aligned
#define WP 72

typedef __attribute__((ext_vector_type(16))) _Float16 v16h;
typedef __attribute__((ext_vector_type(2)))  _Float16 v2h;
typedef __attribute__((ext_vector_type(8)))  float    v8f;

union V16 { uint4 q[2]; v16h v; };

// two f32 -> packed f16 pair (one v_cvt_pkrtz_f16_f32)
__device__ __forceinline__ unsigned int cvt_pk(float a, float b) {
  return __builtin_bit_cast(unsigned int, __builtin_amdgcn_cvt_pkrtz(a, b));
}

// one packed byte (2 nibbles) -> 2 scaled f16 weights, packed in a u32.
// pk halves are 1024+q (exact); (pk - 1032) is Sterbenz-exact (q-8);
// single packed multiply by s gives correctly rounded (q-8)*s.
__device__ __forceinline__ unsigned int dq(int v, v2h s2) {
  unsigned int pk = 0x64006400u | ((unsigned int)v & 15u)
                                | (((unsigned int)v & 0xF0u) << 12);
  v2h h = __builtin_bit_cast(v2h, pk);
  h = (h - (_Float16)1032.0f) * s2;
  return __builtin_bit_cast(unsigned int, h);
}

// stage one K-chunk: x tile (TM x TK f32 -> f16) and w tile (dequant int4)
__device__ __forceinline__ void stage_tiles(
    const float* __restrict__ x, const int* __restrict__ wpk,
    const float* __restrict__ scales,
    unsigned short* bX, unsigned short* bW,
    int tid, int m0, int n0, int kk, int IN, int KG) {
#pragma unroll
  for (int i = 0; i < 8; ++i) {
    int idx = tid + i * 256;            // 0..2047 : 128 rows * 16 float4
    int row = idx >> 4;
    int c4  = idx & 15;
    float4 xv = *(const float4*)(x + (size_t)(m0 + row) * IN + kk + c4 * 4);
    uint2 p;
    p.x = cvt_pk(xv.x, xv.y);
    p.y = cvt_pk(xv.z, xv.w);
    *(uint2*)(bX + row * XP + c4 * 4) = p;
  }
#pragma unroll
  for (int i = 0; i < 4; ++i) {
    int idx = tid + i * 256;            // 0..1023 : 128 rows * 8 int4
    int row = idx >> 3;
    int c4  = idx & 7;
    int4 wv = *(const int4*)(wpk + (size_t)(n0 + row) * (IN >> 1) + (kk >> 1) + c4 * 4);
    float s = scales[(size_t)(n0 + row) * KG + (kk >> 5) + (c4 >> 2)];
    v2h s2 = __builtin_bit_cast(v2h, __builtin_amdgcn_cvt_pkrtz(s, s));
    uint4 o;
    o.x = dq(wv.x, s2);
    o.y = dq(wv.y, s2);
    o.z = dq(wv.z, s2);
    o.w = dq(wv.w, s2);
    *(uint4*)(bW + row * WP + c4 * 8) = o;
  }
}

__global__ __launch_bounds__(256) void qlinear_wmma_f16(
    const float* __restrict__ x,       // [M, IN] f32
    const int*   __restrict__ wpk,     // [OUT, IN/2] int32 (1 byte = 2 nibbles)
    const float* __restrict__ scales,  // [OUT, IN/32] f32
    const float* __restrict__ bias,    // [OUT] f32
    float*       __restrict__ out,     // [M, OUT] f32
    int M, int IN, int OUT) {
  __shared__ unsigned short sX[2][TM * XP];
  __shared__ unsigned short sW[2][TN * WP];

  const int tid  = threadIdx.x;
  const int lane = tid & 31;
  const int wave = tid >> 5;
  const int wr   = wave >> 2;   // 0..1 : wave row (64 M-rows each)
  const int wc   = wave & 3;    // 0..3 : wave col (32 N-cols each)
  const int lr   = lane & 15;   // row within a 16-tile
  const int lh   = lane >> 4;   // half-wave select (WMMA K-half)

  const int m0 = blockIdx.x * TM;
  const int n0 = blockIdx.y * TN;
  const int KG = IN >> 5;       // quant groups per weight row

  v8f acc[4][2] = {};

  float bv[2];
#pragma unroll
  for (int ni = 0; ni < 2; ++ni)
    bv[ni] = bias[n0 + wc * 32 + ni * 16 + lr];

  // prologue: stage first chunk into buffer 0
  stage_tiles(x, wpk, scales, sX[0], sW[0], tid, m0, n0, 0, IN, KG);
  __syncthreads();

  int buf = 0;
  for (int kk = 0; kk < IN; kk += TK) {
    // stage next chunk into the idle buffer while computing on current one
    if (kk + TK < IN)
      stage_tiles(x, wpk, scales, sX[buf ^ 1], sW[buf ^ 1],
                  tid, m0, n0, kk + TK, IN, KG);

    const unsigned short* cX = sX[buf];
    const unsigned short* cW = sW[buf];
#pragma unroll
    for (int ks = 0; ks < 2; ++ks) {
      v16h a[4], b[2];
#pragma unroll
      for (int mi = 0; mi < 4; ++mi) {
        // A 16x32 f16 layout: lane holds row (lane&15); K = lh*8..+8 and +16
        int r  = wr * 64 + mi * 16 + lr;
        int cb = ks * 32 + lh * 8;
        V16 u;
        u.q[0] = *(const uint4*)(cX + r * XP + cb);
        u.q[1] = *(const uint4*)(cX + r * XP + cb + 16);
        a[mi] = u.v;
      }
#pragma unroll
      for (int ni = 0; ni < 2; ++ni) {
        // B 32x16 f16 layout: lane holds col (lane&15); K = lh*8..+8 and +16
        int r  = wc * 32 + ni * 16 + lr;
        int cb = ks * 32 + lh * 8;
        V16 u;
        u.q[0] = *(const uint4*)(cW + r * WP + cb);
        u.q[1] = *(const uint4*)(cW + r * WP + cb + 16);
        b[ni] = u.v;
      }
#pragma unroll
      for (int mi = 0; mi < 4; ++mi)
#pragma unroll
        for (int ni = 0; ni < 2; ++ni)
          acc[mi][ni] = __builtin_amdgcn_wmma_f32_16x16x32_f16(
              false, a[mi], false, b[ni], (short)0, acc[mi][ni], false, false);
    }
    __syncthreads();
    buf ^= 1;
  }

  // epilogue: C/D layout (VGPR j -> M row lh*8+j, N col lane&15)
#pragma unroll
  for (int mi = 0; mi < 4; ++mi) {
#pragma unroll
    for (int ni = 0; ni < 2; ++ni) {
#pragma unroll
      for (int j = 0; j < 8; ++j) {
        int row = m0 + wr * 64 + mi * 16 + lh * 8 + j;
        int col = n0 + wc * 32 + ni * 16 + lr;
        out[(size_t)row * OUT + col] = acc[mi][ni][j] + bv[ni];
      }
    }
  }
}

extern "C" void kernel_launch(void* const* d_in, const int* in_sizes, int n_in,
                              void* d_out, int out_size, void* d_ws, size_t ws_size,
                              hipStream_t stream) {
  const float* x      = (const float*)d_in[0];
  const int*   wpk    = (const int*)d_in[1];
  const float* scales = (const float*)d_in[2];
  const float* bias   = (const float*)d_in[3];
  float*       out    = (float*)d_out;

  const long long OUT = in_sizes[3];                     // bias length
  const long long IN  = (2LL * in_sizes[1]) / OUT;       // w_packed = OUT*IN/2
  const long long M   = (long long)in_sizes[0] / IN;     // x = M*IN

  dim3 grid((unsigned)(M / TM), (unsigned)(OUT / TN));
  qlinear_wmma_f16<<<grid, 256, 0, stream>>>(x, wpk, scales, bias, out,
                                             (int)M, (int)IN, (int)OUT);
}
